// VectorQuantizerEMA_9783935500410
// MI455X (gfx1250) — matrix operand order repaired and checked by reading, not compiled, not run on hardware
//
#include <hip/hip_runtime.h>
#include <hip/hip_bf16.h>

namespace {

constexpr int F = 8;
constexpr int N = 16384;
constexpr int D = 64;
constexpr int K = 512;

typedef __attribute__((ext_vector_type(16))) __bf16 v16bf;
typedef __attribute__((ext_vector_type(8)))  float  v8f;
typedef __attribute__((ext_vector_type(4)))  float  f32x4;
typedef __attribute__((ext_vector_type(4)))  unsigned int u32x4;
typedef int gcc_i32x4 __attribute__((__vector_size__(16)));  // builtin's pointee type

#define AS1 __attribute__((address_space(1)))
#define AS3 __attribute__((address_space(3)))

union BF16Frag { v16bf v; u32x4 q[2]; };

// ---------------------------------------------------------------------------
// Zero scratch (dw + counts + loss accumulator live contiguously in ws).
// ---------------------------------------------------------------------------
__global__ void vq_zero(float* __restrict__ p, int n) {
  const int i = blockIdx.x * blockDim.x + threadIdx.x;
  if (i < n) p[i] = 0.0f;
}

// ---------------------------------------------------------------------------
// Prep: wT[f][k][d] = w[f][d][k] (f32, contiguous codeword rows for gather),
//       wbf same layout in bf16 (WMMA B operand source), wsq[f][k] = ||w_k||^2.
// One thread per (f,k); column reads are coalesced across threads.
// ---------------------------------------------------------------------------
__global__ __launch_bounds__(256) void vq_prep(
    const float* __restrict__ w, __hip_bfloat16* __restrict__ wbf,
    float* __restrict__ wT, float* __restrict__ wsq) {
  const int gid = blockIdx.x * blockDim.x + threadIdx.x;
  if (gid >= F * K) return;
  const int f = gid / K;
  const float* col = w + (size_t)f * D * K + (gid - f * K);
  float* tr = wT + (size_t)gid * D;
  __hip_bfloat16* br = wbf + (size_t)gid * D;
  float s = 0.0f;
#pragma unroll 8
  for (int d = 0; d < D; ++d) {
    const float v = col[(size_t)d * K];
    s += v * v;
    tr[d] = v;
    br[d] = __float2bfloat16(v);
  }
  wsq[gid] = s;
}

// ---------------------------------------------------------------------------
// Argmin GEMM: per block = 128 tokens of one feature, full K=512 codebook
// staged in LDS as bf16 (64KB, async-to-LDS). Each wave: 16 tokens, 32 code
// tiles, 2x v_wmma_f32_16x16x32_bf16 per tile (C = inline 0).
// score = fma(-2, x.w, ||w||^2)   (||x||^2 const per row for argmin).
// ---------------------------------------------------------------------------
__global__ __launch_bounds__(256) void vq_argmin(
    const float* __restrict__ x, const __hip_bfloat16* __restrict__ wbf,
    const float* __restrict__ wsq, int* __restrict__ enc) {
  __shared__ u32x4 sW[(K * D * 2) / 16];  // 4096 * 16B = 64 KB bf16 codebook

  const int tid = threadIdx.x;
  const int f = blockIdx.x >> 7;            // 128 token tiles per feature
  const int tokenBase = (blockIdx.x & 127) << 7;

  // Stage codebook (bf16, [k][d] layout -> 128B per code row).
  const u32x4* gW = (const u32x4*)(wbf + (size_t)f * K * D);
  __builtin_prefetch(gW, 0, 0);             // global_prefetch_b8
#if __has_builtin(__builtin_amdgcn_global_load_async_to_lds_b128)
#pragma unroll 4
  for (int i = tid; i < (K * D * 2) / 16; i += 256) {
    AS1 gcc_i32x4* src = (AS1 gcc_i32x4*)(uintptr_t)(gW + i);  // global addr
    AS3 gcc_i32x4* dst = (AS3 gcc_i32x4*)(uintptr_t)(sW + i);  // LDS offset
    __builtin_amdgcn_global_load_async_to_lds_b128(src, dst, 0, 0);
  }
#if __has_builtin(__builtin_amdgcn_s_wait_asynccnt)
  __builtin_amdgcn_s_wait_asynccnt(0);
#endif
#else
#pragma unroll 4
  for (int i = tid; i < (K * D * 2) / 16; i += 256) sW[i] = gW[i];
#endif
  __syncthreads();

  const int lane = tid & 31;
  const int wave = tid >> 5;
  const int lrow = lane & 15;
  const int half = lane >> 4;
  const int m0 = tokenBase + wave * 16;
  const float* xr  = x + ((size_t)f * N + (m0 + lrow)) * D;
  const float* wsf = wsq + f * K;

  // A fragments (16-bit A 16x32 layout, ISA 7.12.2): lane L (row M=L&15,
  // half=L>>4): v0..v3 hold K=base+half*8..+7 (pairs), v4..v7 hold
  // K=base+16+half*8..+7.  Chunk base: 0 (a0) / 32 (a1).
  BF16Frag a0, a1;
  {
    const int d0 = half * 8;
    f32x4 xa = *(const f32x4*)(xr + d0);
    f32x4 xb = *(const f32x4*)(xr + d0 + 4);
    f32x4 xc = *(const f32x4*)(xr + d0 + 16);
    f32x4 xd = *(const f32x4*)(xr + d0 + 20);
#pragma unroll
    for (int j = 0; j < 4; ++j) {
      a0.v[j]      = (__bf16)xa[j];
      a0.v[4 + j]  = (__bf16)xb[j];
      a0.v[8 + j]  = (__bf16)xc[j];
      a0.v[12 + j] = (__bf16)xd[j];
    }
    xa = *(const f32x4*)(xr + 32 + d0);
    xb = *(const f32x4*)(xr + 32 + d0 + 4);
    xc = *(const f32x4*)(xr + 32 + d0 + 16);
    xd = *(const f32x4*)(xr + 32 + d0 + 20);
#pragma unroll
    for (int j = 0; j < 4; ++j) {
      a1.v[j]      = (__bf16)xa[j];
      a1.v[4 + j]  = (__bf16)xb[j];
      a1.v[8 + j]  = (__bf16)xc[j];
      a1.v[12 + j] = (__bf16)xd[j];
    }
  }

  float best[8];
  int bidx[8];
#pragma unroll
  for (int r = 0; r < 8; ++r) { best[r] = 3.4e38f; bidx[r] = 0; }

  // Unroll 4: four independent WMMA accumulator chains in flight hide the
  // WMMA->WMMA RAW latency and the WMMA->VALU co-exec hazard window; the
  // cmp/cndmask epilogues dual-issue into the remaining slots.
#pragma unroll 4
  for (int kt = 0; kt < K / 16; ++kt) {
    const int c = kt * 16 + lrow;           // this lane's code column
    const u32x4* brow = sW + c * 8;         // 128B bf16 row of code c
    // B fragment (32x16 16-bit): lane holds K rows base+half*8..+7 and
    // base+16+half*8..+7 (matching A) => two 16B chunks of sW[c].
    BF16Frag b0, b1;
    b0.q[0] = brow[half];
    b0.q[1] = brow[2 + half];
    b1.q[0] = brow[4 + half];
    b1.q[1] = brow[6 + half];

    v8f acc = {};                           // SRC2 = inline 0, no reg setup
    acc = __builtin_amdgcn_wmma_f32_16x16x32_bf16(
        false, a0.v, false, b0.v, (short)0, acc, false, false);
    acc = __builtin_amdgcn_wmma_f32_16x16x32_bf16(
        false, a1.v, false, b1.v, (short)0, acc, false, false);

    const float wq = wsf[c];
#pragma unroll
    for (int r = 0; r < 8; ++r) {
      const float score = __builtin_fmaf(-2.0f, acc[r], wq);  // ||w||^2-2x.w
      if (score < best[r]) { best[r] = score; bidx[r] = c; }
    }
  }

  // Reduce argmin across the 16 N-lanes (within each half), first-index ties.
#pragma unroll
  for (int r = 0; r < 8; ++r) {
    for (int off = 8; off; off >>= 1) {
      const float ob = __shfl_xor(best[r], off, 16);
      const int   oi = __shfl_xor(bidx[r], off, 16);
      if (ob < best[r] || (ob == best[r] && oi < bidx[r])) {
        best[r] = ob; bidx[r] = oi;
      }
    }
  }
  if (lrow == 0) {
    int* er = enc + f * N + m0 + half * 8;    // C layout: M = half*8 + r
#pragma unroll
    for (int r = 0; r < 8; ++r) er[r] = bidx[r];
  }
}

// ---------------------------------------------------------------------------
// Gather quantized rows (exact f32 from wT), write quantized_st, loss partials,
// LDS-accumulate dw over two 32-row D halves (64 KB LDS), counts via global
// atomics. Grid: F * 16 blocks, 1024 tokens each.
// ---------------------------------------------------------------------------
__global__ __launch_bounds__(256) void vq_scatter(
    const float* __restrict__ x, const float* __restrict__ wT,
    const int* __restrict__ enc, float* __restrict__ out_q,
    float* __restrict__ dw, float* __restrict__ counts,
    float* __restrict__ lossAcc) {
  __shared__ float dwL[32 * K];               // 64 KB
  const int tid = threadIdx.x;
  const int f = blockIdx.x >> 4;
  const int tb = (blockIdx.x & 15) << 10;

  float lsum = 0.0f;
  for (int h = 0; h < 2; ++h) {
    for (int i = tid; i < 32 * K; i += 256) dwL[i] = 0.0f;
    __syncthreads();

    for (int t = tid; t < 1024; t += 256) {
      const int n = tb + t;
      const int idx = enc[f * N + n];
      const f32x4* xp = (const f32x4*)(x     + ((size_t)f * N + n)   * D + h * 32);
      const f32x4* qp = (const f32x4*)(wT    + ((size_t)f * K + idx) * D + h * 32);
      f32x4*       op = (f32x4*)      (out_q + ((size_t)f * N + n)   * D + h * 32);
#pragma unroll
      for (int j = 0; j < 8; ++j) {
        const f32x4 xv = xp[j];
        const f32x4 qv = qp[j];
        op[j] = qv;                            // quantized_st == quantized
        const f32x4 dv = qv - xv;
        lsum += dv[0] * dv[0] + dv[1] * dv[1] + dv[2] * dv[2] + dv[3] * dv[3];
        const int d = j * 4;
        atomicAdd(&dwL[(d + 0) * K + idx], xv[0]);
        atomicAdd(&dwL[(d + 1) * K + idx], xv[1]);
        atomicAdd(&dwL[(d + 2) * K + idx], xv[2]);
        atomicAdd(&dwL[(d + 3) * K + idx], xv[3]);
      }
      if (h == 0) atomicAdd(&counts[f * K + idx], 1.0f);
    }
    __syncthreads();

    for (int i = tid; i < 32 * K; i += 256) {
      const float v = dwL[i];
      if (v != 0.0f) atomicAdd(&dw[((size_t)f * D + h * 32) * K + i], v);
    }
    __syncthreads();
  }

  for (int off = 16; off; off >>= 1) lsum += __shfl_down(lsum, off, 32);
  if ((tid & 31) == 0) atomicAdd(lossAcc, lsum);
}

// ---------------------------------------------------------------------------
// Finalize: EMA cluster sizes, Laplace smoothing, new_ema_w, new_w, loss.
// One block per feature; thread k owns code k.
// ---------------------------------------------------------------------------
__global__ __launch_bounds__(512) void vq_finalize(
    const float* __restrict__ ema_cs, const float* __restrict__ ema_w,
    const float* __restrict__ counts, const float* __restrict__ dw,
    const float* __restrict__ lossAcc, float* __restrict__ out_loss,
    float* __restrict__ out_w, float* __restrict__ out_cs,
    float* __restrict__ out_ema_w) {
  __shared__ float red[K];
  const int f = blockIdx.x;
  const int k = threadIdx.x;

  const float cs = 0.99f * ema_cs[f * K + k] + 0.01f * counts[f * K + k];
  out_cs[f * K + k] = cs;
  red[k] = cs;
  __syncthreads();
  for (int off = K / 2; off; off >>= 1) {
    if (k < off) red[k] += red[k + off];
    __syncthreads();
  }
  const float nsum = red[0];
  const float smoothed = (cs + 1e-5f) / (nsum + (float)K * 1e-5f) * nsum;
  const float inv = 1.0f / smoothed;

  for (int d = 0; d < D; ++d) {
    const size_t i = ((size_t)f * D + d) * K + k;
    const float ne = 0.99f * ema_w[i] + 0.01f * dw[i];
    out_ema_w[i] = ne;
    out_w[i] = ne * inv;
  }
  if (f == 0 && k == 0)
    out_loss[0] = 0.25f * lossAcc[0] / (float)((size_t)F * N * D);
}

}  // namespace

// ---------------------------------------------------------------------------
// Host launcher.
// ---------------------------------------------------------------------------
extern "C" void kernel_launch(void* const* d_in, const int* in_sizes, int n_in,
                              void* d_out, int out_size, void* d_ws,
                              size_t ws_size, hipStream_t stream) {
  (void)in_sizes; (void)n_in; (void)out_size; (void)ws_size;

  const float* x      = (const float*)d_in[0];  // [F,N,D]
  const float* w      = (const float*)d_in[1];  // [F,D,K]
  const float* ema_cs = (const float*)d_in[2];  // [F,K]
  const float* ema_w  = (const float*)d_in[3];  // [F,D,K]

  float* out = (float*)d_out;
  // Flat output layout: quantized_st | loss | new_w | new_cluster_size | new_ema_w
  constexpr size_t Q_OFF    = 0;
  constexpr size_t LOSS_OFF = (size_t)F * N * D;            // 8388608
  constexpr size_t W_OFF    = LOSS_OFF + 1;                 // 8388609
  constexpr size_t CS_OFF   = W_OFF + (size_t)F * D * K;    // 8650753
  constexpr size_t EMA_OFF  = CS_OFF + (size_t)F * K;       // 8654849
  float* out_q    = out + Q_OFF;
  float* out_loss = out + LOSS_OFF;
  float* out_w    = out + W_OFF;
  float* out_cs   = out + CS_OFF;
  float* out_ema  = out + EMA_OFF;

  // Workspace layout (bytes).
  char* ws = (char*)d_ws;
  __hip_bfloat16* wbf  = (__hip_bfloat16*)(ws + 0);        //  524288 B [f][k][d] bf16
  float* wT            = (float*)(ws + 524288);            // 1048576 B [f][k][d] f32
  float* wsq           = (float*)(ws + 1572864);           //   16384 B [f][k]
  int*   enc           = (int*)  (ws + 1589248);           //  524288 B [f][n]
  float* dw            = (float*)(ws + 2113536);           // 1048576 B [f][d][k]
  float* counts        = (float*)(ws + 3162112);           //   16384 B [f][k]
  float* lossAcc       = (float*)(ws + 3178496);           //       4 B
  // dw..lossAcc are contiguous: zero in one pass.
  constexpr int ZERO_N = F * D * K + F * K + 1;            // 266241 floats

  vq_zero<<<(ZERO_N + 255) / 256, 256, 0, stream>>>(dw, ZERO_N);
  vq_prep<<<(F * K + 255) / 256, 256, 0, stream>>>(w, wbf, wT, wsq);
  vq_argmin<<<F * (N / 128), 256, 0, stream>>>(x, wbf, wsq, enc);
  vq_scatter<<<F * 16, 256, 0, stream>>>(x, wT, enc, out_q, dw, counts, lossAcc);
  vq_finalize<<<F, 512, 0, stream>>>(ema_cs, ema_w, counts, dw, lossAcc,
                                     out_loss, out_w, out_cs, out_ema);
}
